// TransformerEncoderLayerWithROPE_65704409694881
// MI455X (gfx1250) — matrix operand order repaired and checked by reading, not compiled, not run on hardware
//
#include <hip/hip_runtime.h>
#include <hip/hip_bf16.h>

// ---------------------------------------------------------------------------
// CDNA5 (gfx1250) transformer encoder layer with RoPE.
// GEMMs + attention on v_wmma_f32_16x16x32_bf16 (bf16 in, f32 accum), with
// async global->LDS staging (ASYNCcnt) and double-buffered LDS tiles.
// Workspace requirement: ~248 MiB (see offsets in kernel_launch).
// ---------------------------------------------------------------------------

typedef __attribute__((ext_vector_type(16))) __bf16 v16bf;
typedef __attribute__((ext_vector_type(8)))  float  v8f;

union FragAB { uint4 u[2]; v16bf v; };

__device__ __forceinline__ unsigned short f2bf(float f) {
    union { float f; unsigned u; } v; v.f = f;
    unsigned r = v.u + 0x7FFFu + ((v.u >> 16) & 1u);   // round-to-nearest-even
    return (unsigned short)(r >> 16);
}

// Async DMA: global memory -> LDS, 16 bytes per lane. Tracked by ASYNCcnt.
// lds = 32-bit LDS byte address (low 32 bits of generic shared pointer).
__device__ __forceinline__ void async_load_b128(unsigned lds,
                                                const unsigned short* g) {
    asm volatile("global_load_async_to_lds_b128 %0, %1, off"
                 :: "v"(lds), "v"(g) : "memory");
}
__device__ __forceinline__ void wait_async0() {
    asm volatile("s_wait_asynccnt 0" ::: "memory");
}
__device__ __forceinline__ unsigned lds_addr(const void* p) {
    return (unsigned)(uintptr_t)p;   // flat->LDS: low 32 bits are the offset
}

// ---------------------------------------------------------------------------
// Elementwise casts / reshapes
// ---------------------------------------------------------------------------
__global__ void cast_bf16_kernel(const float* __restrict__ in,
                                 unsigned short* __restrict__ out, int n) {
    int i = blockIdx.x * 256 + threadIdx.x;
    if (i < n) out[i] = f2bf(in[i]);
}

// in: [K,N] f32 row-major  ->  out: [N,K] bf16 row-major ("Bt" layout)
__global__ void tcast_bf16_kernel(const float* __restrict__ in,
                                  unsigned short* __restrict__ out,
                                  int K, int N) {
    int i = blockIdx.x * 256 + threadIdx.x;
    if (i < K * N) {
        int k = i % K, n = i / K;
        out[(size_t)n * K + k] = f2bf(in[(size_t)k * N + n]);
    }
}

// RoPE: in fp32 [B,S,1024] (projected q or k) -> out bf16 [B,H,S,64] rotated.
__global__ void rope_kernel(const float* __restrict__ in,
                            unsigned short* __restrict__ out, int S) {
    int idx = blockIdx.x * 256 + threadIdx.x;   // B*H*S*32 threads
    int j = idx & 31; idx >>= 5;
    int s = idx % S;  idx /= S;
    int h = idx & 15; int b = idx >> 4;
    size_t ib = ((size_t)b * S + s) * 1024 + h * 64;
    float t0 = in[ib + 2 * j];
    float t1 = in[ib + 2 * j + 1];
    int ic = (2 * j < 32)     ? 2 * j     : 2 * j - 32;
    int is = (2 * j + 1 < 32) ? 2 * j + 1 : 2 * j + 1 - 32;
    float ac = (float)s * __powf(10000.f, -(float)ic / 32.f);
    float as = (float)s * __powf(10000.f, -(float)is / 32.f);
    float c = __cosf(ac), sn = __sinf(as);
    size_t ob = (((size_t)(b * 16 + h) * S) + s) * 64;
    out[ob + j]      = f2bf(t0 * c - t1 * sn);
    out[ob + j + 32] = f2bf(t0 * sn + t1 * c);
}

// V: fp32 [B,S,1024] -> bf16 [B,H,64,S]  (transposed per head -> "Bt" layout)
__global__ void reshape_v_kernel(const float* __restrict__ in,
                                 unsigned short* __restrict__ out, int S) {
    int idx = blockIdx.x * 256 + threadIdx.x;   // B*H*64*S threads
    int s = idx % S; idx /= S;
    int d = idx & 63; idx >>= 6;
    int h = idx & 15; int b = idx >> 4;
    out[(((size_t)(b * 16 + h) * 64) + d) * S + s] =
        f2bf(in[((size_t)b * S + s) * 1024 + h * 64 + d]);
}

// out = LayerNorm(X + Y) * g + be ; optional bf16 mirror for next GEMM's A.
__global__ __launch_bounds__(256) void add_ln_kernel(
    const float* __restrict__ X, const float* __restrict__ Y,
    const float* __restrict__ g, const float* __restrict__ be,
    float* __restrict__ outF, unsigned short* __restrict__ outB) {
    __shared__ float red[256];
    int tid = threadIdx.x;
    size_t row = blockIdx.x;
    const float* xr = X + row * 1024;
    const float* yr = Y + row * 1024;
    float v[4]; float sum = 0.f;
    #pragma unroll
    for (int i = 0; i < 4; i++) {
        v[i] = xr[tid + i * 256] + yr[tid + i * 256];
        sum += v[i];
    }
    red[tid] = sum; __syncthreads();
    for (int st = 128; st > 0; st >>= 1) {
        if (tid < st) red[tid] += red[tid + st];
        __syncthreads();
    }
    float mu = red[0] * (1.f / 1024.f);
    __syncthreads();
    float vs = 0.f;
    #pragma unroll
    for (int i = 0; i < 4; i++) { float d = v[i] - mu; vs += d * d; }
    red[tid] = vs; __syncthreads();
    for (int st = 128; st > 0; st >>= 1) {
        if (tid < st) red[tid] += red[tid + st];
        __syncthreads();
    }
    float rstd = rsqrtf(red[0] * (1.f / 1024.f) + 1e-5f);
    #pragma unroll
    for (int i = 0; i < 4; i++) {
        int col = tid + i * 256;
        float o = (v[i] - mu) * rstd * g[col] + be[col];
        outF[row * 1024 + col] = o;
        if (outB) outB[row * 1024 + col] = f2bf(o);
    }
}

// ---------------------------------------------------------------------------
// Generic bf16 WMMA GEMM:  C[M,N] = A[M,K] * Bt[N,K]^T + bias   (TN form)
// Block tile 128x128, BK=64, 8 waves (4x2), wave tile 32x64 (2x4 WMMA accums).
// Double-buffered LDS, async global->LDS staging, one barrier per K-step.
// ---------------------------------------------------------------------------
#define TM 128
#define TN 128
#define TK 64

__global__ __launch_bounds__(256) void gemm_bf16_tn(
    const unsigned short* __restrict__ A,   // [M,K] bf16
    const unsigned short* __restrict__ Bt,  // [N,K] bf16
    const float* __restrict__ bias,         // [N] or null
    float* __restrict__ outF,               // [M,N] f32 or null
    unsigned short* __restrict__ outB,      // [M,N] bf16 or null
    int M, int N, int K, int relu) {
    __shared__ __align__(16) unsigned short As[2][TM * TK];
    __shared__ __align__(16) unsigned short Bs[2][TN * TK];
    int tid  = threadIdx.x;
    int wave = tid >> 5, lane = tid & 31;
    int wm = wave >> 1, wn = wave & 1;
    int half = lane >> 4, l15 = lane & 15;
    int bm = blockIdx.y * TM, bn = blockIdx.x * TN;

    v8f acc[2][4] = {};

    // Async-stage one 128x64 A tile + 128x64 B tile (1024 chunks each).
    auto stage = [&](int buf, int k0) {
        #pragma unroll
        for (int it = 0; it < 4; it++) {
            int c = tid + it * 256;
            int r = c >> 3, cc = c & 7;
            async_load_b128(lds_addr(&As[buf][r * TK + cc * 8]),
                            &A[(size_t)(bm + r) * K + k0 + cc * 8]);
            async_load_b128(lds_addr(&Bs[buf][r * TK + cc * 8]),
                            &Bt[(size_t)(bn + r) * K + k0 + cc * 8]);
        }
    };

    stage(0, 0);
    int nk = K / TK;
    for (int ki = 0; ki < nk; ki++) {
        wait_async0();            // this wave's tile-ki DMAs have landed
        __syncthreads();          // => everyone's have; prior reads done
        if (ki + 1 < nk) stage((ki + 1) & 1, (ki + 1) * TK);

        const unsigned short* Ab = As[ki & 1];
        const unsigned short* Bb = Bs[ki & 1];
        #pragma unroll
        for (int ks = 0; ks < 2; ks++) {
            // A fragments: regs0-3 = K(half*8..+7), regs4-7 = K(16+half*8..+7)
            v16bf af[2];
            #pragma unroll
            for (int m = 0; m < 2; m++) {
                int row = wm * 32 + m * 16 + l15;
                FragAB t;
                t.u[0] = *(const uint4*)&Ab[row * TK + ks * 32 + half * 8];
                t.u[1] = *(const uint4*)&Ab[row * TK + ks * 32 + 16 + half * 8];
                af[m] = t.v;
            }
            // B fragments: regs0-3 = K(half*8..+7), regs4-7 = K(16+half*8..+7)
            v16bf bf[4];
            #pragma unroll
            for (int n = 0; n < 4; n++) {
                int row = wn * 64 + n * 16 + l15;
                FragAB t;
                t.u[0] = *(const uint4*)&Bb[row * TK + ks * 32 + half * 8];
                t.u[1] = *(const uint4*)&Bb[row * TK + ks * 32 + 16 + half * 8];
                bf[n] = t.v;
            }
            #pragma unroll
            for (int m = 0; m < 2; m++)
                #pragma unroll
                for (int n = 0; n < 4; n++)
                    acc[m][n] = __builtin_amdgcn_wmma_f32_16x16x32_bf16(
                        false, af[m], false, bf[n], (short)0, acc[m][n],
                        false, false);
        }
    }

    // Epilogue (C layout: M = r + 8*half, N = lane&15 per 16x16 tile)
    #pragma unroll
    for (int m = 0; m < 2; m++)
        #pragma unroll
        for (int n = 0; n < 4; n++) {
            int col = bn + wn * 64 + n * 16 + l15;
            float bv = bias ? bias[col] : 0.f;
            #pragma unroll
            for (int r = 0; r < 8; r++) {
                int row = bm + wm * 32 + m * 16 + r + half * 8;
                float v = acc[m][n][r] + bv;
                if (relu) v = v > 0.f ? v : 0.f;
                if (outF) outF[(size_t)row * N + col] = v;
                if (outB) outB[(size_t)row * N + col] = f2bf(v);
            }
        }
}

// ---------------------------------------------------------------------------
// Fused flash-style attention, one block per (b*H, 128 q rows).
// Qr,Kr: bf16 [B*H, S, 64] (RoPE'd); Vt: bf16 [B*H, 64, S]; Ctx: bf16 [B,S,1024]
// K/V tiles double-buffered + async staged; P relayout through wave-private LDS
// (DS ops from one wave are in-order => no barrier needed between store/load).
// ---------------------------------------------------------------------------
__global__ __launch_bounds__(256) void attn_kernel(
    const unsigned short* __restrict__ Qr,
    const unsigned short* __restrict__ Kr,
    const unsigned short* __restrict__ Vt,
    unsigned short* __restrict__ Ctx, int S) {
    __shared__ __align__(16) unsigned short Qs[128 * 64];
    __shared__ __align__(16) unsigned short Ks[2][64 * 64];
    __shared__ __align__(16) unsigned short Vs[2][64 * 64];
    __shared__ __align__(16) unsigned short Ps[8 * 16 * 64];

    int tid  = threadIdx.x;
    int wave = tid >> 5, lane = tid & 31;
    int half = lane >> 4, l15 = lane & 15;
    int bh = blockIdx.y;                 // b*16 + h
    int q0 = blockIdx.x * 128;

    const unsigned short* Qbase = Qr + (size_t)bh * S * 64;
    const unsigned short* Kbase = Kr + (size_t)bh * S * 64;
    const unsigned short* Vbase = Vt + (size_t)bh * 64 * S;

    auto stageKV = [&](int buf, int kt) {
        #pragma unroll
        for (int it = 0; it < 2; it++) {
            int c = tid + it * 256;
            int r = c >> 3, cc = c & 7;
            async_load_b128(lds_addr(&Ks[buf][r * 64 + cc * 8]),
                            &Kbase[(size_t)(kt + r) * 64 + cc * 8]);
            async_load_b128(lds_addr(&Vs[buf][r * 64 + cc * 8]),
                            &Vbase[(size_t)r * S + kt + cc * 8]);
        }
    };

    // Stage Q tile (128x64 = 1024 chunks) and first K/V tiles.
    #pragma unroll
    for (int it = 0; it < 4; it++) {
        int c = tid + it * 256;
        int r = c >> 3, cc = c & 7;
        async_load_b128(lds_addr(&Qs[r * 64 + cc * 8]),
                        &Qbase[(size_t)(q0 + r) * 64 + cc * 8]);
    }
    stageKV(0, 0);
    wait_async0();
    __syncthreads();

    // Per-wave Q fragments (persist across k-tiles)
    v16bf aq[2];
    #pragma unroll
    for (int ks = 0; ks < 2; ks++) {
        int row = wave * 16 + l15;
        FragAB t;
        t.u[0] = *(const uint4*)&Qs[row * 64 + ks * 32 + half * 8];
        t.u[1] = *(const uint4*)&Qs[row * 64 + ks * 32 + 16 + half * 8];
        aq[ks] = t.v;
    }

    v8f cacc[4] = {};
    float mrow[8], lrow[8];
    #pragma unroll
    for (int r = 0; r < 8; r++) { mrow[r] = -1e30f; lrow[r] = 0.f; }

    unsigned short* Pw = &Ps[wave * 16 * 64];
    int ntile = S / 64;

    for (int it = 0; it < ntile; it++) {
        if (it + 1 < ntile) stageKV((it + 1) & 1, (it + 1) * 64);
        const unsigned short* Kb = Ks[it & 1];
        const unsigned short* Vb = Vs[it & 1];

        // logits tile: 16 q rows x 64 k cols (4 WMMA tiles, Dh=64 -> 2 steps)
        v8f sacc[4] = {};
        #pragma unroll
        for (int n = 0; n < 4; n++) {
            int krow = n * 16 + l15;
            #pragma unroll
            for (int ks = 0; ks < 2; ks++) {
                FragAB t;
                t.u[0] = *(const uint4*)&Kb[krow * 64 + ks * 32 + half * 8];
                t.u[1] = *(const uint4*)&Kb[krow * 64 + ks * 32 + 16 + half * 8];
                sacc[n] = __builtin_amdgcn_wmma_f32_16x16x32_bf16(
                    false, aq[ks], false, t.v, (short)0, sacc[n], false, false);
            }
        }

        // Online softmax (row = r + 8*half lives across the 16-lane group)
        const float sc = 0.125f;   // 1/sqrt(64)
        #pragma unroll
        for (int r = 0; r < 8; r++) {
            float mx = -1e30f;
            #pragma unroll
            for (int n = 0; n < 4; n++) mx = fmaxf(mx, sacc[n][r]);
            mx *= sc;
            #pragma unroll
            for (int off = 8; off >= 1; off >>= 1)
                mx = fmaxf(mx, __shfl_xor(mx, off, 32));
            float mnew = fmaxf(mrow[r], mx);
            float resc = __expf(mrow[r] - mnew);
            float psum = 0.f;
            #pragma unroll
            for (int n = 0; n < 4; n++) {
                float p = __expf(sacc[n][r] * sc - mnew);
                sacc[n][r] = p;
                psum += p;
            }
            #pragma unroll
            for (int off = 8; off >= 1; off >>= 1)
                psum += __shfl_xor(psum, off, 32);
            lrow[r] = lrow[r] * resc + psum;
            mrow[r] = mnew;
            #pragma unroll
            for (int n = 0; n < 4; n++) cacc[n][r] *= resc;
        }

        // C-layout -> A-fragment relayout of P through wave-private LDS.
        #pragma unroll
        for (int n = 0; n < 4; n++)
            #pragma unroll
            for (int r = 0; r < 8; r++)
                Pw[(r + half * 8) * 64 + n * 16 + l15] = f2bf(sacc[n][r]);

        // ctx += P(16x64) @ V(64x64)
        #pragma unroll
        for (int ks = 0; ks < 2; ks++) {
            FragAB tp;
            tp.u[0] = *(const uint4*)&Pw[l15 * 64 + ks * 32 + half * 8];
            tp.u[1] = *(const uint4*)&Pw[l15 * 64 + ks * 32 + 16 + half * 8];
            v16bf ap = tp.v;
            #pragma unroll
            for (int n = 0; n < 4; n++) {
                int vrow = n * 16 + l15;
                FragAB tv;
                tv.u[0] = *(const uint4*)&Vb[vrow * 64 + ks * 32 + half * 8];
                tv.u[1] = *(const uint4*)&Vb[vrow * 64 + ks * 32 + 16 + half * 8];
                cacc[n] = __builtin_amdgcn_wmma_f32_16x16x32_bf16(
                    false, ap, false, tv.v, (short)0, cacc[n], false, false);
            }
        }

        wait_async0();            // next tile landed (this wave)
        __syncthreads();          // => all waves; this tile's reads all done
    }

    // Normalize and scatter to Ctx [B,S,1024] (bf16)
    int b = bh >> 4, h = bh & 15;
    #pragma unroll
    for (int r = 0; r < 8; r++) {
        float inv = 1.f / lrow[r];
        int srow = q0 + wave * 16 + r + half * 8;
        #pragma unroll
        for (int n = 0; n < 4; n++) {
            int d = n * 16 + l15;
            Ctx[((size_t)b * S + srow) * 1024 + h * 64 + d] =
                f2bf(cacc[n][r] * inv);
        }
    }
}

// ---------------------------------------------------------------------------
// Launch sequence
// ---------------------------------------------------------------------------
extern "C" void kernel_launch(void* const* d_in, const int* in_sizes, int n_in,
                              void* d_out, int out_size, void* d_ws,
                              size_t ws_size, hipStream_t stream) {
    const float* x   = (const float*)d_in[0];
    const float* wq  = (const float*)d_in[1];
    const float* bq  = (const float*)d_in[2];
    const float* wk  = (const float*)d_in[3];
    const float* bk  = (const float*)d_in[4];
    const float* wv  = (const float*)d_in[5];
    const float* bv  = (const float*)d_in[6];
    const float* wo  = (const float*)d_in[7];
    const float* bo  = (const float*)d_in[8];
    const float* w1  = (const float*)d_in[9];
    const float* b1  = (const float*)d_in[10];
    const float* w2  = (const float*)d_in[11];
    const float* b2  = (const float*)d_in[12];
    const float* g1  = (const float*)d_in[13];
    const float* be1 = (const float*)d_in[14];
    const float* g2  = (const float*)d_in[15];
    const float* be2 = (const float*)d_in[16];
    float* out = (float*)d_out;
    char* ws = (char*)d_ws;

    const int B = 4, S = 2048, D = 1024, H = 16, FF = 4096;
    const int M = B * S;                        // 8192

    // Workspace layout (bytes). Peak ~248 MiB; aliased regions noted.
    const size_t OFF_WQT = 0;                                 // 2 MiB
    const size_t OFF_WKT = OFF_WQT + (size_t)D * D * 2;
    const size_t OFF_WVT = OFF_WKT + (size_t)D * D * 2;
    const size_t OFF_WOT = OFF_WVT + (size_t)D * D * 2;
    const size_t OFF_W1T = OFF_WOT + (size_t)D * D * 2;       // 8 MiB
    const size_t OFF_W2T = OFF_W1T + (size_t)D * FF * 2;      // 8 MiB
    const size_t OFF_XB  = OFF_W2T + (size_t)FF * D * 2;      // 16 MiB
    const size_t OFF_QF  = OFF_XB  + (size_t)M * D * 2;       // 32 MiB f32
    const size_t OFF_KF  = OFF_QF  + (size_t)M * D * 4;
    const size_t OFF_VF  = OFF_KF  + (size_t)M * D * 4;
    const size_t OFF_QR  = OFF_VF  + (size_t)M * D * 4;       // 16 MiB bf16
    const size_t OFF_KR  = OFF_QR  + (size_t)M * D * 2;
    const size_t OFF_VT  = OFF_KR  + (size_t)M * D * 2;
    const size_t OFF_FF1 = OFF_VT  + (size_t)M * D * 2;       // 64 MiB bf16
    // Aliases (lifetimes do not overlap):
    const size_t OFF_CTX = OFF_QF;                            // bf16, over QF
    const size_t OFF_AO  = OFF_CTX + (size_t)M * D * 2;       // f32
    const size_t OFF_H   = OFF_AO  + (size_t)M * D * 4;       // f32
    const size_t OFF_HB  = OFF_H   + (size_t)M * D * 4;       // bf16
    const size_t OFF_FF2 = OFF_QR;                            // f32, over QR/KR

    auto U16 = [&](size_t o) { return (unsigned short*)(ws + o); };
    auto F32 = [&](size_t o) { return (float*)(ws + o); };

    // --- precision conversion & weight pre-transpose (to Bt = [N][K]) ---
    cast_bf16_kernel<<<(M * D) / 256, 256, 0, stream>>>(x, U16(OFF_XB), M * D);
    tcast_bf16_kernel<<<(D * D) / 256, 256, 0, stream>>>(wq, U16(OFF_WQT), D, D);
    tcast_bf16_kernel<<<(D * D) / 256, 256, 0, stream>>>(wk, U16(OFF_WKT), D, D);
    tcast_bf16_kernel<<<(D * D) / 256, 256, 0, stream>>>(wv, U16(OFF_WVT), D, D);
    tcast_bf16_kernel<<<(D * D) / 256, 256, 0, stream>>>(wo, U16(OFF_WOT), D, D);
    tcast_bf16_kernel<<<(D * FF) / 256, 256, 0, stream>>>(w1, U16(OFF_W1T), D, FF);
    tcast_bf16_kernel<<<(FF * D) / 256, 256, 0, stream>>>(w2, U16(OFF_W2T), FF, D);

    // --- QKV projections ---
    dim3 gD(D / TN, M / TM);
    gemm_bf16_tn<<<gD, 256, 0, stream>>>(U16(OFF_XB), U16(OFF_WQT), bq,
                                         F32(OFF_QF), nullptr, M, D, D, 0);
    gemm_bf16_tn<<<gD, 256, 0, stream>>>(U16(OFF_XB), U16(OFF_WKT), bk,
                                         F32(OFF_KF), nullptr, M, D, D, 0);
    gemm_bf16_tn<<<gD, 256, 0, stream>>>(U16(OFF_XB), U16(OFF_WVT), bv,
                                         F32(OFF_VF), nullptr, M, D, D, 0);

    // --- RoPE + head reshapes ---
    rope_kernel<<<(B * H * S * 32) / 256, 256, 0, stream>>>(F32(OFF_QF),
                                                            U16(OFF_QR), S);
    rope_kernel<<<(B * H * S * 32) / 256, 256, 0, stream>>>(F32(OFF_KF),
                                                            U16(OFF_KR), S);
    reshape_v_kernel<<<(B * H * 64 * S) / 256, 256, 0, stream>>>(F32(OFF_VF),
                                                                 U16(OFF_VT), S);

    // --- fused attention ---
    dim3 gA(S / 128, B * H);
    attn_kernel<<<gA, 256, 0, stream>>>(U16(OFF_QR), U16(OFF_KR), U16(OFF_VT),
                                        U16(OFF_CTX), S);

    // --- output projection + residual LN1 ---
    gemm_bf16_tn<<<gD, 256, 0, stream>>>(U16(OFF_CTX), U16(OFF_WOT), bo,
                                         F32(OFF_AO), nullptr, M, D, D, 0);
    add_ln_kernel<<<M, 256, 0, stream>>>(x, F32(OFF_AO), g1, be1,
                                         F32(OFF_H), U16(OFF_HB));

    // --- FFN ---
    dim3 gF(FF / TN, M / TM);
    gemm_bf16_tn<<<gF, 256, 0, stream>>>(U16(OFF_HB), U16(OFF_W1T), b1,
                                         nullptr, U16(OFF_FF1), M, FF, D, 1);
    gemm_bf16_tn<<<gD, 256, 0, stream>>>(U16(OFF_FF1), U16(OFF_W2T), b2,
                                         F32(OFF_FF2), nullptr, M, D, FF, 0);

    // --- residual LN2 -> final output (fp32) ---
    add_ln_kernel<<<M, 256, 0, stream>>>(F32(OFF_H), F32(OFF_FF2), g2, be2,
                                         out, nullptr);
}
